// FakeNewsNet_90031104459403
// MI455X (gfx1250) — compile-verified, matrix operand order checked by reading, not run on hardware
//
#include <hip/hip_runtime.h>
#include <hip/hip_bf16.h>
#include <math.h>

typedef __attribute__((ext_vector_type(16))) __bf16 v16bf;
typedef __attribute__((ext_vector_type(8)))  float  v8f;

#define B_  256
#define S_  512
#define D_  300
#define H_  128
#define KP_ 320   // D padded up to a multiple of 32 for the K=32 WMMA depth

// ---- workspace layout ----
// xp is stored TRANSPOSED for the scan: xp_t[t][b][h]  (contiguous 8KB tile per
// (t, 16-batch-row) group -> async-DMA friendly)
#define XP_BYTES  ((size_t)B_ * S_ * H_ * 4)        // 64 MiB
#define WIH_OFF   (XP_BYTES)
#define WIH_BYTES ((size_t)H_ * KP_ * 2)            // bf16 W_ih, [h][kp], zero-padded K
#define WHH_OFF   (WIH_OFF + WIH_BYTES)
#define WHH_BYTES ((size_t)H_ * H_ * 2)             // bf16 W_hh^T, [k][n] = W_hh[n][k]

// ------------------------------------------------------------------
// Prep: fp32 weights -> bf16 workspace copies (padded / transposed)
// ------------------------------------------------------------------
__global__ __launch_bounds__(256)
void prep_kernel(const float* __restrict__ W_ih,   // [H][D]
                 const float* __restrict__ W_hh,   // [H][H]
                 __bf16* __restrict__ wihBf,       // [H][KP]
                 __bf16* __restrict__ whhT)        // [H][H] k-major
{
    int tid = blockIdx.x * blockDim.x + threadIdx.x;
    const int nWih = H_ * KP_;
    if (tid < nWih) {
        int h = tid / KP_, k = tid % KP_;
        float v = (k < D_) ? W_ih[h * D_ + k] : 0.0f;
        wihBf[tid] = (__bf16)v;
    } else {
        int t = tid - nWih;
        if (t < H_ * H_) {
            int k = t / H_, n = t % H_;
            whhT[t] = (__bf16)W_hh[n * H_ + k];   // B-matrix layout: [k][n]
        }
    }
}

// ------------------------------------------------------------------
// Kernel 1: embedding gather + input projection GEMM (bf16 WMMA)
//   xp_t[t][b][n] = sum_k emb[x[b*S+t]][k] * W_ih[n][k] + b_ih[n]
// One block per 16-row M tile; 8 waves = 8 N tiles of 16.
// ------------------------------------------------------------------
__global__ __launch_bounds__(256)
void emb_proj_kernel(const int*   __restrict__ x,      // [B*S]
                     const float* __restrict__ emb,    // [V][D]
                     const __bf16* __restrict__ wihBf, // [H][KP]
                     const float* __restrict__ b_ih,   // [H]
                     float* __restrict__ xpT)          // [S][B][H]
{
    __shared__ __bf16 aLds[16][KP_];   // 10 KB: gathered+padded A tile, bf16
    __shared__ int    tok[16];

    const int mtile = blockIdx.x;      // 0 .. B*S/16 - 1
    const int tid   = threadIdx.x;

    if (tid < 16) tok[tid] = x[mtile * 16 + tid];
    __syncthreads();

    // prefetch the 16 embedding rows (L2-resident table, hide gather latency)
    {
        const int prow = tid >> 4, pseg = tid & 15;
        __builtin_prefetch(emb + (size_t)tok[prow] * D_ + pseg * 20, 0, 3);
    }

    // cooperative gather of 16 embedding rows (convert fp32 -> bf16, pad K)
    for (int e = tid; e < 16 * KP_; e += 256) {
        int row = e / KP_, col = e % KP_;
        float v = (col < D_) ? emb[(size_t)tok[row] * D_ + col] : 0.0f;
        aLds[row][col] = (__bf16)v;
    }
    __syncthreads();

    const int lane  = tid & 31;
    const int ntile = tid >> 5;          // wave id = N tile
    const int half  = lane >> 4;         // 0: lanes 0-15, 1: lanes 16-31
    const int mrow  = lane & 15;
    const int ncol  = ntile * 16 + (lane & 15);
    const int kbase = half * 8;          // ISA 16-bit A/B fragment K pattern

    v8f acc = {};
    for (int kc = 0; kc < KP_ / 32; ++kc) {
        v16bf a, b;
        #pragma unroll
        for (int j = 0; j < 8; ++j) {
            const int k0 = kc * 32 + kbase + j;        // K = kbase + j
            const int k1 = kc * 32 + kbase + 16 + j;   // K = kbase + 16 + j
            a[j]     = aLds[mrow][k0];
            a[j + 8] = aLds[mrow][k1];
            b[j]     = wihBf[(size_t)ncol * KP_ + k0];
            b[j + 8] = wihBf[(size_t)ncol * KP_ + k1];
        }
        acc = __builtin_amdgcn_wmma_f32_16x16x32_bf16(
                  false, a, false, b, (short)0, acc, false, false);
    }

    const float bias = b_ih[ncol];
    #pragma unroll
    for (int r = 0; r < 8; ++r) {
        const int m = mtile * 16 + half * 8 + r;       // C/D layout: M = half*8 + r
        const int bb = m >> 9;                          // m / S_
        const int tt = m & (S_ - 1);                    // m % S_
        xpT[((size_t)tt * B_ + bb) * H_ + ncol] = acc[r] + bias;
    }
}

// ------------------------------------------------------------------
// Kernel 2: recurrent scan + fused max/mean pooling + FC head.
// Batch rows independent -> 16 blocks of 16 batch rows each.
// h state (16x128) lives in LDS as bf16; W_hh^T staged in LDS.
// Per-step x-tile (8KB, contiguous) is double-buffered into LDS with
// global_load_async_to_lds_b128 one step ahead (ASYNCcnt pipelining).
// ------------------------------------------------------------------
__global__ __launch_bounds__(256)
void rnn_kernel(const float*  __restrict__ xpT,   // [S][B][H]
                const __bf16* __restrict__ whhTg, // [H][H] k-major
                const float*  __restrict__ b_hh,  // [H]
                const float*  __restrict__ fc_w,  // [2H]
                const float*  __restrict__ fc_b,  // [1]
                float* __restrict__ out)          // [B]
{
    __shared__ __align__(16) __bf16 whh[H_][H_];      // 32 KB, [k][n]; reused as pooled[] after scan
    __shared__ __align__(16) __bf16 hBf[16][H_];      //  4 KB, current h in bf16
    __shared__ __align__(16) float  xtile[2][16][H_]; // 16 KB, double-buffered x_t tile

    const int tid = threadIdx.x;
    const int b0  = blockIdx.x * 16;

    // LDS byte offsets for the async-DMA destinations (flat LDS addr = low 32 bits)
    const unsigned xl[2] = { (unsigned)(unsigned long long)&xtile[0][0][0],
                             (unsigned)(unsigned long long)&xtile[1][0][0] };

    // prologue: async-stage x-tile for t=0 into buffer 0 (overlaps weight staging)
    {
        const unsigned long long ga =
            (unsigned long long)(xpT + (size_t)b0 * H_) + (unsigned)tid * 16u;
        const unsigned l = xl[0] + (unsigned)tid * 16u;
        asm volatile("global_load_async_to_lds_b128 %0, %1, off"
                     :: "v"(l), "v"(ga) : "memory");
        asm volatile("global_load_async_to_lds_b128 %0, %1, off"
                     :: "v"(l + 4096u), "v"(ga + 4096ull) : "memory");
    }

    __bf16* whhp = &whh[0][0];
    __bf16* hp   = &hBf[0][0];
    for (int e = tid; e < H_ * H_; e += 256) whhp[e] = whhTg[e];
    for (int e = tid; e < 16 * H_;  e += 256) hp[e]  = (__bf16)0.0f;

    const int lane  = tid & 31;
    const int ntile = tid >> 5;
    const int half  = lane >> 4;
    const int mrow  = lane & 15;
    const int ncol  = ntile * 16 + (lane & 15);
    const int kbase = half * 8;
    const float bias = b_hh[ncol];

    v8f maxacc, sumacc;
    #pragma unroll
    for (int r = 0; r < 8; ++r) { maxacc[r] = -1e30f; sumacc[r] = 0.0f; }

    __syncthreads();   // whh / hBf staged

    for (int t = 0; t < S_; ++t) {
        const int buf = t & 1;

        // (a) issue async DMA for step t+1 into the other buffer
        if (t + 1 < S_) {
            const unsigned long long ga =
                (unsigned long long)(xpT + ((size_t)(t + 1) * B_ + b0) * H_)
                + (unsigned)tid * 16u;
            const unsigned l = xl[buf ^ 1] + (unsigned)tid * 16u;
            asm volatile("global_load_async_to_lds_b128 %0, %1, off"
                         :: "v"(l), "v"(ga) : "memory");
            asm volatile("global_load_async_to_lds_b128 %0, %1, off"
                         :: "v"(l + 4096u), "v"(ga + 4096ull) : "memory");
        }

        // (b) h @ W_hh^T : 4 K-chunks of 32, A from LDS h, B from LDS W_hh^T
        v8f acc = {};
        #pragma unroll
        for (int kc = 0; kc < H_ / 32; ++kc) {
            v16bf a, b;
            #pragma unroll
            for (int j = 0; j < 8; ++j) {
                const int k0 = kc * 32 + kbase + j;
                const int k1 = kc * 32 + kbase + 16 + j;
                a[j]     = hBf[mrow][k0];
                a[j + 8] = hBf[mrow][k1];
                b[j]     = whh[k0][ncol];
                b[j + 8] = whh[k1][ncol];
            }
            acc = __builtin_amdgcn_wmma_f32_16x16x32_bf16(
                      false, a, false, b, (short)0, acc, false, false);
        }

        // (c) this step's x-tile DMA complete (async loads retire in order:
        //     <=2 outstanding leaves only the just-issued t+1 copies in flight)
        if (t + 1 < S_) asm volatile("s_wait_asynccnt 0x2" ::: "memory");
        else            asm volatile("s_wait_asynccnt 0x0" ::: "memory");
        __syncthreads();   // (d) x-tile visible to all; hBf reads of (b) done

        // (e) combine + tanh, fused pooling accumulation, write new h
        #pragma unroll
        for (int r = 0; r < 8; ++r) {
            float v = acc[r] + xtile[buf][half * 8 + r][ncol] + bias;
            v = tanhf(v);
            maxacc[r]  = fmaxf(maxacc[r], v);
            sumacc[r] += v;
            hBf[half * 8 + r][ncol] = (__bf16)v;
        }
        __syncthreads();   // (f) new h visible before next step's reads
    }

    // pooling results -> LDS (reuse the dead whh region), then tiny FC head
    float* pooled = (float*)&whh[0][0];     // 16 KB needed of the 32 KB region
    #pragma unroll
    for (int r = 0; r < 8; ++r) {
        const int br = half * 8 + r;
        pooled[br * (2 * H_) + ncol]      = maxacc[r];
        pooled[br * (2 * H_) + H_ + ncol] = sumacc[r] * (1.0f / (float)S_);
    }
    __syncthreads();

    if (tid < 16) {
        float s = fc_b[0];
        for (int j = 0; j < 2 * H_; ++j) s += pooled[tid * (2 * H_) + j] * fc_w[j];
        out[b0 + tid] = s;
    }
}

// ------------------------------------------------------------------
extern "C" void kernel_launch(void* const* d_in, const int* in_sizes, int n_in,
                              void* d_out, int out_size, void* d_ws, size_t ws_size,
                              hipStream_t stream) {
    (void)in_sizes; (void)n_in; (void)out_size; (void)ws_size;

    const int*   x    = (const int*)  d_in[0];
    const float* emb  = (const float*)d_in[1];
    const float* W_ih = (const float*)d_in[2];
    const float* W_hh = (const float*)d_in[3];
    const float* b_ih = (const float*)d_in[4];
    const float* b_hh = (const float*)d_in[5];
    const float* fc_w = (const float*)d_in[6];
    const float* fc_b = (const float*)d_in[7];
    float* out = (float*)d_out;

    char*   ws    = (char*)d_ws;
    float*  xpT   = (float*) (ws);             // 64 MiB, [S][B][H]
    __bf16* wihBf = (__bf16*)(ws + WIH_OFF);   // 80 KB
    __bf16* whhT  = (__bf16*)(ws + WHH_OFF);   //  32 KB

    const int prepElems = H_ * KP_ + H_ * H_;
    prep_kernel<<<(prepElems + 255) / 256, 256, 0, stream>>>(W_ih, W_hh, wihBf, whhT);

    emb_proj_kernel<<<(B_ * S_) / 16, 256, 0, stream>>>(x, emb, wihBf, b_ih, xpT);

    rnn_kernel<<<B_ / 16, 256, 0, stream>>>(xpT, whhT, b_hh, fc_w, fc_b, out);
}
